// CRF_69501160784720
// MI455X (gfx1250) — matrix operand order repaired and checked by reading, not compiled, not run on hardware
//
#include <hip/hip_runtime.h>

// CRF loss: B=256, T=512, C=128.
// Scan kernel: 16 blocks x 256 threads (8 wave32). Each block owns 16 batch
// rows; per time step it does a [16x128]x[128x128] GEMM on the WMMA path
// (bf16x3 split for ~f32 accuracy) to realize the log-sum-exp recursion.
// Emissions are prefetched 2 steps ahead via GLOBAL_LOAD_ASYNC_TO_LDS_B128
// (ASYNCcnt-tracked).

#define Bsz 256
#define Tsz 512
#define Csz 128
#define ROWS 16
#define NBLK (Bsz / ROWS)

typedef __attribute__((ext_vector_type(16))) __bf16 v16bf;
typedef __attribute__((ext_vector_type(8)))  float  v8f;
typedef int v4i __attribute__((vector_size(16)));   // matches builtin param type

#if defined(__has_builtin)
#if __has_builtin(__builtin_amdgcn_global_load_async_to_lds_b128)
#define HAVE_ASYNC 1
#endif
#endif
#ifndef HAVE_ASYNC
#define HAVE_ASYNC 0
#endif

#if HAVE_ASYNC
#define ASYNC_LD16(gptr, lptr, off)                                        \
  __builtin_amdgcn_global_load_async_to_lds_b128(                          \
      (__attribute__((address_space(1))) v4i*)(gptr),                      \
      (__attribute__((address_space(3))) v4i*)(lptr), (off), 0)
#if __has_builtin(__builtin_amdgcn_s_wait_asynccnt)
#define WAIT_ASYNC(n) __builtin_amdgcn_s_wait_asynccnt(n)
#else
#define WAIT_ASYNC(n) asm volatile("s_wait_asynccnt %0" ::"i"(n) : "memory")
#endif
#endif

// ---- fast transcendentals: raw v_exp_f32 / v_log_f32, no denorm fixups ----
// exp inputs are <= 0 (row-max normalized): underflow flush-to-zero is the
// desired logsumexp behavior. log inputs are >= ~0.9: never denormal.
#if defined(__has_builtin) && __has_builtin(__builtin_amdgcn_exp2f)
__device__ __forceinline__ float fexp(float x) {
  return __builtin_amdgcn_exp2f(x * 1.44269504088896340736f);
}
#else
__device__ __forceinline__ float fexp(float x) { return __expf(x); }
#endif
#if defined(__has_builtin) && __has_builtin(__builtin_amdgcn_logf)
__device__ __forceinline__ float flog(float x) {
  return __builtin_amdgcn_logf(x) * 0.69314718055994530942f;
}
#else
__device__ __forceinline__ float flog(float x) { return __logf(x); }
#endif

// ---- monotone float <-> uint key (for deterministic integer atomic max) ----
__device__ __forceinline__ unsigned f32_key(float f) {
  unsigned b = __float_as_uint(f);
  return (b & 0x80000000u) ? ~b : (b | 0x80000000u);
}
__device__ __forceinline__ float key_f32(unsigned k) {
  return __uint_as_float((k & 0x80000000u) ? (k & 0x7FFFFFFFu) : ~k);
}

// ---- bf16 helpers (RNE), bit-level to avoid relying on __bf16 arithmetic ----
__device__ __forceinline__ unsigned short f32_bf16_bits(float f) {
  unsigned b = __float_as_uint(f);
  return (unsigned short)((b + 0x7FFFu + ((b >> 16) & 1u)) >> 16);
}
__device__ __forceinline__ float bf16_bits_f32(unsigned short h) {
  return __uint_as_float(((unsigned)h) << 16);
}
__device__ __forceinline__ __bf16 bf16_from_bits(unsigned short u) {
  union { unsigned short s; __bf16 h; } x; x.s = u; return x.h;
}

__global__ __launch_bounds__(256) void crf_forward_scan(
    const float* __restrict__ em,          // [B,T,C]
    const unsigned char* __restrict__ mask,// [B,T] (bool bytes, prefix-true)
    const float* __restrict__ trans,       // [C,C]
    float* __restrict__ log_den)           // [B]
{
  __shared__ __align__(32) float    alpha[ROWS][Csz];    // 8 KB: current log_alpha
  __shared__ __align__(32) float    emitb[3][ROWS][Csz]; // 24 KB: emissions pipeline
  __shared__ __align__(32) unsigned short aAhi[4 * 32 * 16]; // 4 KB: A operand hi
  __shared__ __align__(32) unsigned short aAlo[4 * 32 * 16]; // 4 KB: A operand lo
  __shared__ unsigned rowmaxU[2][ROWS];                  // double-buffered row maxes
  __shared__ unsigned lenLDS[ROWS];                      // per-row sequence lengths
  __shared__ float    redbuf[ROWS][16];                  // epilogue partial sums

  const int tid  = threadIdx.x;
  const int lane = tid & 31;
  const int wave = tid >> 5;              // j-tile id: columns [wave*16, wave*16+16)
  const int b0   = blockIdx.x * ROWS;
  const int m    = tid >> 4;              // row 0..15 (16 threads per row)
  const int col0 = (tid & 15) * 8;        // 8 columns per thread

  // ---------------- E = exp(transitions), bf16 hi/lo, in B-operand layout ----
  // B matrix (32x16, KxN) in this code's contiguous k order (A matches): lanes 0-15 hold N=lane, K=k0..k0+15;
  // lanes 16-31 hold N=lane-16, K=k0+16..k0+31.
  v16bf Bhi[4], Blo[4];
  {
    const int n = lane & 15, khalf = lane >> 4, jb = wave * 16;
#pragma unroll
    for (int c = 0; c < 4; ++c) {
      v16bf bh, bl;
#pragma unroll
      for (int e = 0; e < 16; ++e) {
        int K = c * 32 + khalf * 16 + e;
        float v = fexp(trans[K * Csz + jb + n]);
        unsigned short h = f32_bf16_bits(v);
        bh[e] = bf16_from_bits(h);
        bl[e] = bf16_from_bits(f32_bf16_bits(v - bf16_bits_f32(h)));
      }
      Bhi[c] = bh; Blo[c] = bl;
    }
  }

  // ---------------- init: lengths, alpha = emissions[:,0,:], row maxes ------
  if (tid < ROWS) { lenLDS[tid] = 0u; rowmaxU[0][tid] = 0u; rowmaxU[1][tid] = 0u; }
  __syncthreads();
  {
    const unsigned char* mrow = mask + (size_t)(b0 + m) * Tsz + (tid & 15) * 32;
    unsigned cnt = 0;
#pragma unroll 8
    for (int u = 0; u < 32; ++u) cnt += mrow[u];
    atomicAdd(&lenLDS[m], cnt);

    const float* e0 = em + (size_t)(b0 + m) * Tsz * Csz + col0;
    float lmax = -3.0e38f;
#pragma unroll
    for (int i = 0; i < 8; ++i) {
      float v = e0[i];
      alpha[m][col0 + i] = v;
      lmax = fmaxf(lmax, v);
    }
    atomicMax(&rowmaxU[0][m], f32_key(lmax));
  }
  __syncthreads();

  // ---------------- emissions prefetch pipeline (depth 2) -------------------
  const float* emrow = em + (size_t)(b0 + m) * Tsz * Csz + col0;
#if HAVE_ASYNC
  // 3 LDS buffers; async DMA global->LDS, 2 x b128 per thread per step.
  ASYNC_LD16(emrow + 1 * Csz, &emitb[1][m][col0], 0);
  ASYNC_LD16(emrow + 1 * Csz, &emitb[1][m][col0], 16);
  ASYNC_LD16(emrow + 2 * Csz, &emitb[2][m][col0], 0);
  ASYNC_LD16(emrow + 2 * Csz, &emitb[2][m][col0], 16);
#else
  float4 pa0 = *(const float4*)(emrow + 1 * Csz);
  float4 pb0 = *(const float4*)(emrow + 1 * Csz + 4);
  float4 pa1 = *(const float4*)(emrow + 2 * Csz);
  float4 pb1 = *(const float4*)(emrow + 2 * Csz + 4);
#endif

  int cur = 0;
  int buf = 1;    // t % 3, rotating
  int bufp2 = 0;  // (t+2) % 3, rotating
  for (int t = 1; t < Tsz; ++t) {
    const int nxt = cur ^ 1;

    // ======== producer phase ========
#if HAVE_ASYNC
    // Loads complete in order: waiting for <=2 outstanding guarantees step-t
    // data is in LDS (2 pairs in flight: t, t+1). Then launch t+2.
    if (t + 1 < Tsz) { WAIT_ASYNC(2); } else { WAIT_ASYNC(0); }
    if (t + 2 < Tsz) {
      ASYNC_LD16(emrow + (size_t)(t + 2) * Csz, &emitb[bufp2][m][col0], 0);
      ASYNC_LD16(emrow + (size_t)(t + 2) * Csz, &emitb[bufp2][m][col0], 16);
    }
#else
    if (((t - 1) & 1) == 0) {
      *(float4*)&emitb[buf][m][col0]     = pa0;
      *(float4*)&emitb[buf][m][col0 + 4] = pb0;
      if (t + 2 < Tsz) {
        pa0 = *(const float4*)(emrow + (size_t)(t + 2) * Csz);
        pb0 = *(const float4*)(emrow + (size_t)(t + 2) * Csz + 4);
      }
    } else {
      *(float4*)&emitb[buf][m][col0]     = pa1;
      *(float4*)&emitb[buf][m][col0 + 4] = pb1;
      if (t + 2 < Tsz) {
        pa1 = *(const float4*)(emrow + (size_t)(t + 2) * Csz);
        pb1 = *(const float4*)(emrow + (size_t)(t + 2) * Csz + 4);
      }
    }
#endif
    if (tid < ROWS) rowmaxU[nxt][tid] = 0u;

    // a_hat = exp(alpha - rowmax), bf16 hi/lo, scattered into A-operand layout:
    // A (16x32, MxK): lane = m + 16*((k>>4)&1); elem = k&15.
    {
      const float rm = key_f32(rowmaxU[cur][m]);
#pragma unroll
      for (int i = 0; i < 8; ++i) {
        int k  = col0 + i;
        int c  = k >> 5, kk = k & 31;
        int la = m + 16 * ((kk >> 4) & 1);
        int e  = kk & 15;
        float v = fexp(alpha[m][k] - rm);
        unsigned short h = f32_bf16_bits(v);
        int off = (c * 32 + la) * 16 + e;
        aAhi[off] = h;
        aAlo[off] = f32_bf16_bits(v - bf16_bits_f32(h));
      }
    }
    __syncthreads();

    // ======== WMMA phase: D = a_hat * E  (bf16x3 -> ~f32) ========
    v16bf Ah[4], Al[4];
#pragma unroll
    for (int c = 0; c < 4; ++c) {            // issue all A loads as one clause
      Ah[c] = *(const v16bf*)&aAhi[(c * 32 + lane) * 16];
      Al[c] = *(const v16bf*)&aAlo[(c * 32 + lane) * 16];
    }
    v8f acc = {};
#pragma unroll
    for (int c = 0; c < 4; ++c) {
      acc = __builtin_amdgcn_wmma_f32_16x16x32_bf16(false, Ah[c], false, Bhi[c],
                                                    (short)0, acc, false, false);
      acc = __builtin_amdgcn_wmma_f32_16x16x32_bf16(false, Ah[c], false, Blo[c],
                                                    (short)0, acc, false, false);
      acc = __builtin_amdgcn_wmma_f32_16x16x32_bf16(false, Al[c], false, Bhi[c],
                                                    (short)0, acc, false, false);
    }

    // epilogue: new alpha = log(acc) + rowmax + emit, masked per row.
    // D layout: lanes 0-15: N=lane, M=r; lanes 16-31: N=lane-16, M=8+r.
    {
      const int nloc = lane & 15, hi = lane >> 4;
      const int j = wave * 16 + nloc;
#pragma unroll
      for (int r = 0; r < 8; ++r) {
        int md = r + 8 * hi;
        float nv = flog(acc[r]) + key_f32(rowmaxU[cur][md]) + emitb[buf][md][j];
        float ov = alpha[md][j];
        float out = (t < (int)lenLDS[md]) ? nv : ov;
        alpha[md][j] = out;
        atomicMax(&rowmaxU[nxt][md], f32_key(out));
      }
    }
    __syncthreads();
    cur = nxt;
    buf   = (buf   == 2) ? 0 : buf + 1;
    bufp2 = (bufp2 == 2) ? 0 : bufp2 + 1;
  }

  // ---------------- log_den[b] = rowmax + log(sum exp(alpha - rowmax)) ------
  {
    const float rm = key_f32(rowmaxU[cur][m]);
    float s = 0.f;
#pragma unroll
    for (int i = 0; i < 8; ++i) s += fexp(alpha[m][col0 + i] - rm);
    redbuf[m][tid & 15] = s;
  }
  __syncthreads();
  if (tid < ROWS) {
    float tot = 0.f;
#pragma unroll
    for (int k = 0; k < 16; ++k) tot += redbuf[tid][k];   // fixed order: deterministic
    log_den[b0 + tid] = key_f32(rowmaxU[cur][tid]) + flog(tot);
  }
}

// Numerator (gold-path score) + final mean. One block, one thread per batch row.
__global__ __launch_bounds__(256) void crf_num_and_reduce(
    const float* __restrict__ em, const int* __restrict__ tags,
    const unsigned char* __restrict__ mask, const float* __restrict__ trans,
    const float* __restrict__ log_den, float* __restrict__ out)
{
  const int b = threadIdx.x;
  const int* tg = tags + (size_t)b * Tsz;
  const unsigned char* mk = mask + (size_t)b * Tsz;
  const float* eb = em + (size_t)b * Tsz * Csz;

  float ln = 0.f;
  int pt = 0, pm = 0;
  for (int t = 0; t < Tsz; ++t) {
    int tgt = tg[t];
    int mt  = mk[t] ? 1 : 0;
    if (mt) ln += eb[(size_t)t * Csz + tgt];
    if (t > 0 && mt && pm) ln += trans[pt * Csz + tgt];
    pt = tgt; pm = mt;
  }

  __shared__ float red[256];
  red[b] = log_den[b] - ln;
  __syncthreads();
#pragma unroll
  for (int s = 128; s > 0; s >>= 1) {   // fixed tree: deterministic
    if (b < s) red[b] += red[b + s];
    __syncthreads();
  }
  if (b == 0) out[0] = red[0] / (float)Bsz;
}

extern "C" void kernel_launch(void* const* d_in, const int* in_sizes, int n_in,
                              void* d_out, int out_size, void* d_ws, size_t ws_size,
                              hipStream_t stream) {
  (void)in_sizes; (void)n_in; (void)out_size; (void)ws_size;
  const float*         em    = (const float*)d_in[0];          // [B,T,C] f32
  const int*           tags  = (const int*)d_in[1];            // [B,T] i32
  const unsigned char* mask  = (const unsigned char*)d_in[2];  // [B,T] bool
  const float*         trans = (const float*)d_in[3];          // [C,C] f32
  float* log_den = (float*)d_ws;                               // [B] scratch
  float* outp    = (float*)d_out;                              // scalar

  crf_forward_scan<<<dim3(NBLK), dim3(256), 0, stream>>>(em, mask, trans, log_den);
  crf_num_and_reduce<<<dim3(1), dim3(256), 0, stream>>>(em, tags, mask, trans, log_den, outp);
}
